// Model_51419348467864
// MI455X (gfx1250) — matrix operand order repaired and checked by reading, not compiled, not run on hardware
//
#include <hip/hip_runtime.h>
#include <math.h>

#define Bn      64
#define Ln      512
#define Cn      321
#define NCn     8
#define Dn      64
#define PREDn   96
#define Mrows   (Bn * Cn)          // 20544, divisible by 16
#define NORMF   0.125f             // 1/sqrt(64)

typedef __attribute__((ext_vector_type(16))) __bf16 v16bf;
typedef __attribute__((ext_vector_type(8)))  __bf16 v8bf;
typedef __attribute__((ext_vector_type(8)))  float  v8f;

// ---- bf16 conversion: native casts; clang lowers to v_cvt_pk_bf16_f32 where available
static __device__ __forceinline__ __bf16 f2bf(float f) { return (__bf16)f; }

// Swap bits 3 and 4 of an index (within-32-chunk swizzle). With this storage
// order, an A fragment's 16 lane elements are CONTIGUOUS: positions
// [0..15] of a chunk hold K={0..7,16..23} (half 0), [16..31] hold
// K={8..15,24..31} (half 1) -- matching ISA 7.12.2 16-bit A layout.
static __device__ __forceinline__ int sw34(int i) {
    return (i & ~24) | ((i & 8) << 1) | ((i & 16) >> 1);
}

// One 32B load serves both A (swizzled buffers) and B (swizzled N-major rows).
static __device__ __forceinline__ v16bf ld_frag(const __bf16* row, int kc, int lane) {
    return *(const v16bf*)(row + kc + ((lane >> 4) << 4));
}
// A fragment built from linear fp32 (embedding rows): vector loads + packed cvt.
static __device__ __forceinline__ v16bf cvt_a_frag_f32(const float* row, int kc, int lane) {
    const float* p = row + kc + ((lane >> 4) << 3);
    v8f lo = *(const v8f*)p;          // K = koff .. koff+7
    v8f hi = *(const v8f*)(p + 16);   // K = koff+16 .. koff+23
    v8bf blo = __builtin_convertvector(lo, v8bf);
    v8bf bhi = __builtin_convertvector(hi, v8bf);
    return __builtin_shufflevector(blo, bhi, 0, 1, 2, 3, 4, 5, 6, 7,
                                   8, 9, 10, 11, 12, 13, 14, 15);
}
static __device__ __forceinline__ v8f wmma_bf16(v16bf a, v16bf b, v8f c) {
    return __builtin_amdgcn_wmma_f32_16x16x32_bf16(false, a, false, b, (short)0, c, false, false);
}

// ---------------- P0: weight fold + bf16 convert (incl. Wq/Wk/Wv), zero scalar
__global__ void k_wprep(const float* __restrict__ Ws, const float* __restrict__ Wt,
                        const float* __restrict__ Wg,
                        const float* __restrict__ Wq, const float* __restrict__ Wk,
                        const float* __restrict__ Wv,
                        __bf16* __restrict__ Bs16, __bf16* __restrict__ Btm16,
                        __bf16* __restrict__ Bg16, __bf16* __restrict__ Bst16,
                        __bf16* __restrict__ Bgr16, __bf16* __restrict__ Bqkv16,
                        float* __restrict__ out) {
    int tid = blockIdx.x * blockDim.x + threadIdx.x;
    if (tid == 0) out[(size_t)2 * Bn * PREDn * Cn] = 0.0f;   // third output: zeros((1,))
    if (tid < 3 * Dn * Dn) {                                 // QKV weights, swizzled bf16
        int w = tid >> 12, off = tid & 4095;
        const float* src = (w == 0) ? Wq : (w == 1) ? Wk : Wv;
        Bqkv16[sw34(tid)] = f2bf(src[off]);
    }
    if (tid >= PREDn * 704) return;
    int p = tid / 704, j = tid % 704;
    const float* wsr = Ws + p * 1216;
    const float* wtr = Wt + p * 1216;
    const float* wgr = Wg + p * 1216;
    Bst16[p * 704 + sw34(j)] = f2bf(wsr[512 + j] + wtr[512 + j]);  // (Ws+Wt) tail
    Bgr16[p * 704 + sw34(j)] = f2bf(wgr[512 + j]);
    if (j < 512) {
        Bs16[p * 512 + sw34(j)]  = f2bf(wsr[j]);
        Btm16[p * 512 + sw34(j)] = f2bf(wtr[j] - wsr[j]);          // (Wt-Ws) head
        Bg16[p * 512 + sw34(j)]  = f2bf(wgr[j]);
    }
}

// ---------------- P1: moving-average decomposition, writes swizzled A layout
__global__ void k_decomp(const float* __restrict__ x_enc,
                         __bf16* __restrict__ Xbf, __bf16* __restrict__ Tbf) {
    int b = blockIdx.x;
    int c = blockIdx.y * blockDim.x + threadIdx.x;
    if (c >= Cn) return;
    const float* base = x_enc + (size_t)b * Ln * Cn + c;
    int m = b * Cn + c;
    __bf16* xr = Xbf + (size_t)m * Ln;
    __bf16* tr = Tbf + (size_t)m * Ln;
    float S = 12.0f * base[0];                 // 12 left-pad copies of x[0]
    for (int j = 0; j <= 12; ++j) S += base[j * Cn];
    const float inv = 1.0f / 25.0f;
    for (int t = 0; t < Ln; ++t) {
        int pos = sw34(t);
        xr[pos] = f2bf(base[t * Cn]);
        tr[pos] = f2bf(S * inv);
        int ja = t + 13; if (ja > Ln - 1) ja = Ln - 1;
        int jr = t - 12; if (jr < 0) jr = 0;
        S += base[ja * Cn] - base[jr * Cn];
    }
}

// ---------------- P2: embeddings + WMMA QKV + softmax attention + time features
// One wave handles two (b,c) pairs: 16 A-rows = 2 pairs x 8 embeddings.
__global__ void __launch_bounds__(32) k_attn_tf(
    const int* __restrict__ idx, const float* __restrict__ emb,
    const int* __restrict__ x_dec,
    const float* __restrict__ te0, const float* __restrict__ te1, const float* __restrict__ te2,
    const __bf16* __restrict__ Bqkv16,
    const float* __restrict__ bq, const float* __restrict__ bk, const float* __restrict__ bv,
    __bf16* __restrict__ Rbf) {
    __shared__ float Qs[16 * 64], Ks[16 * 64], Vs[16 * 64];
    int lane = threadIdx.x;
    int pair0 = blockIdx.x * 2;
    int row = lane & 15;
    int half = lane >> 4;
    int pr_row = row >> 3, n_row = row & 7;
    int m_row = pair0 + pr_row;
    int b_row = m_row / Cn, c_row = m_row % Cn;
    int e_idx = idx[(b_row * NCn + n_row) * Cn + c_row];
    const float* erow = emb + ((size_t)n_row * 1000 + e_idx) * Dn;
    v16bf a0 = cvt_a_frag_f32(erow, 0, lane);
    v16bf a1 = cvt_a_frag_f32(erow, 32, lane);

    const float* B3[3] = {bq, bk, bv};
    float* O3[3] = {Qs, Ks, Vs};
#pragma unroll
    for (int w = 0; w < 3; ++w) {
        const __bf16* Wb = Bqkv16 + w * (Dn * Dn);
        const float* bias = B3[w];
        float* O = O3[w];
#pragma unroll
        for (int t = 0; t < 4; ++t) {                 // N tiles of 16 over h=64
            const __bf16* wrow = Wb + (t * 16 + row) * Dn;
            v8f acc = {};
            acc = wmma_bf16(a0, ld_frag(wrow, 0, lane), acc);
            acc = wmma_bf16(a1, ld_frag(wrow, 32, lane), acc);
            float bvv = bias[t * 16 + row];
#pragma unroll
            for (int r = 0; r < 8; ++r)               // D layout: M = r + 8*half, N = row
                O[(r + 8 * half) * 64 + t * 16 + row] = acc[r] + bvv;
        }
    }
    __syncthreads();

    if (lane < 16) {                                  // per-pair 8x8 softmax attention
        int pr = lane >> 3, n = lane & 7;
        const float* qrow = Qs + (pr * 8 + n) * 64;
        float sc[8];
        float mx = -1e30f;
#pragma unroll
        for (int m2 = 0; m2 < 8; ++m2) {
            const float* krow = Ks + (pr * 8 + m2) * 64;
            float s = 0.f;
            for (int h = 0; h < 64; ++h) s += qrow[h] * krow[h];
            sc[m2] = s;
            mx = fmaxf(mx, s);
        }
        float sum = 0.f;
#pragma unroll
        for (int m2 = 0; m2 < 8; ++m2) { sc[m2] = __expf(sc[m2] - mx); sum += sc[m2]; }
        float scale = NORMF / sum;
        int mrow = pair0 + pr;
        __bf16* rrow = Rbf + (size_t)mrow * 704;
        int jbase = 192 + n * 64;
        for (int d = 0; d < 64; ++d) {
            float o = 0.f;
#pragma unroll
            for (int m2 = 0; m2 < 8; ++m2) o += sc[m2] * Vs[(pr * 8 + m2) * 64 + d];
            rrow[sw34(jbase + d)] = f2bf(o * scale);
        }
    }
    for (int v = lane; v < 384; v += 32) {            // time-feature embeddings (192/pair)
        int pr = v / 192, j = v % 192;
        int mrow = pair0 + pr;
        int bb = mrow / Cn, cc = mrow % Cn;
        const int* xd = x_dec + (bb * Cn + cc) * 3;
        int tbl = j >> 6, d = j & 63;
        float val = (tbl == 0) ? te0[xd[0] * Dn + d]
                  : (tbl == 1) ? te1[xd[1] * Dn + d]
                               : te2[xd[2] * Dn + d];
        Rbf[(size_t)mrow * 704 + sw34(j)] = f2bf(val);
    }
}

// ---------------- G: fused triple linear head; one wave = 16 rows x 32 cols
__global__ void __launch_bounds__(32) k_gemm(
    const __bf16* __restrict__ Xbf, const __bf16* __restrict__ Tbf, const __bf16* __restrict__ Rbf,
    const __bf16* __restrict__ Bs16, const __bf16* __restrict__ Btm16, const __bf16* __restrict__ Bg16,
    const __bf16* __restrict__ Bst16, const __bf16* __restrict__ Bgr16,
    const float* __restrict__ bs, const float* __restrict__ bt, const float* __restrict__ bgv,
    float* __restrict__ out) {
    int lane = threadIdx.x;
    int mt = blockIdx.x;
    int row = lane & 15;
    int half = lane >> 4;
    int g_arow = mt * 16 + row;
    const __bf16* xr = Xbf + (size_t)g_arow * 512;
    const __bf16* tr = Tbf + (size_t)g_arow * 512;
    const __bf16* rr = Rbf + (size_t)g_arow * 704;
    int p0 = blockIdx.y * 32 + row;                   // two adjacent N tiles
    int p1 = p0 + 16;
    const __bf16* bs0 = Bs16 + (size_t)p0 * 512;  const __bf16* bs1 = Bs16 + (size_t)p1 * 512;
    const __bf16* btm0 = Btm16 + (size_t)p0 * 512; const __bf16* btm1 = Btm16 + (size_t)p1 * 512;
    const __bf16* bg0 = Bg16 + (size_t)p0 * 512;  const __bf16* bg1 = Bg16 + (size_t)p1 * 512;
    const __bf16* bst0 = Bst16 + (size_t)p0 * 704; const __bf16* bst1 = Bst16 + (size_t)p1 * 704;
    const __bf16* bgr0 = Bgr16 + (size_t)p0 * 704; const __bf16* bgr1 = Bgr16 + (size_t)p1 * 704;

    v8f aM0 = {}, aM1 = {}, aS0 = {}, aS1 = {};
#pragma unroll 4
    for (int kc = 0; kc < 512; kc += 32) {            // mu = Ws*x + (Wt-Ws)*trend ; sig = Wg*x
        v16bf ax = ld_frag(xr, kc, lane);
        v16bf at = ld_frag(tr, kc, lane);
        aM0 = wmma_bf16(ax, ld_frag(bs0, kc, lane), aM0);
        aM1 = wmma_bf16(ax, ld_frag(bs1, kc, lane), aM1);
        aM0 = wmma_bf16(at, ld_frag(btm0, kc, lane), aM0);
        aM1 = wmma_bf16(at, ld_frag(btm1, kc, lane), aM1);
        aS0 = wmma_bf16(ax, ld_frag(bg0, kc, lane), aS0);
        aS1 = wmma_bf16(ax, ld_frag(bg1, kc, lane), aS1);
    }
#pragma unroll 2
    for (int kc = 0; kc < 704; kc += 32) {            // shared [tf | attn_out] tail
        v16bf ar = ld_frag(rr, kc, lane);
        aM0 = wmma_bf16(ar, ld_frag(bst0, kc, lane), aM0);
        aM1 = wmma_bf16(ar, ld_frag(bst1, kc, lane), aM1);
        aS0 = wmma_bf16(ar, ld_frag(bgr0, kc, lane), aS0);
        aS1 = wmma_bf16(ar, ld_frag(bgr1, kc, lane), aS1);
    }
    float bm0 = bs[p0] + bt[p0], bm1 = bs[p1] + bt[p1];
    float bg0v = bgv[p0], bg1v = bgv[p1];
    const size_t SOFF = (size_t)Bn * PREDn * Cn;
#pragma unroll
    for (int r = 0; r < 8; ++r) {
        int M = r + 8 * half;
        int g = mt * 16 + M;
        int bb = g / Cn, cc = g % Cn;
        size_t rowoff = (size_t)bb * PREDn * Cn + cc;
        float mu0 = aM0[r] + bm0;
        float mu1 = aM1[r] + bm1;
        float z0 = aS0[r] + bg0v;
        float z1 = aS1[r] + bg1v;
        float sp0 = fmaxf(z0, 0.f) + log1pf(__expf(-fabsf(z0)));
        float sp1 = fmaxf(z1, 0.f) + log1pf(__expf(-fabsf(z1)));
        out[rowoff + (size_t)p0 * Cn] = mu0;
        out[rowoff + (size_t)p1 * Cn] = mu1;
        out[SOFF + rowoff + (size_t)p0 * Cn] = sp0;
        out[SOFF + rowoff + (size_t)p1 * Cn] = sp1;
    }
}

extern "C" void kernel_launch(void* const* d_in, const int* in_sizes, int n_in,
                              void* d_out, int out_size, void* d_ws, size_t ws_size,
                              hipStream_t stream) {
    (void)in_sizes; (void)n_in; (void)out_size; (void)ws_size;
    const float* x_enc = (const float*)d_in[0];
    const int*   x_dec = (const int*)d_in[1];
    const int*   idx   = (const int*)d_in[2];
    const float* emb   = (const float*)d_in[3];
    const float* te0   = (const float*)d_in[4];
    const float* te1   = (const float*)d_in[5];
    const float* te2   = (const float*)d_in[6];
    const float* Wq = (const float*)d_in[7];  const float* bq = (const float*)d_in[8];
    const float* Wk = (const float*)d_in[9];  const float* bk = (const float*)d_in[10];
    const float* Wv = (const float*)d_in[11]; const float* bv = (const float*)d_in[12];
    const float* Ws = (const float*)d_in[13]; const float* bs = (const float*)d_in[14];
    const float* Wt = (const float*)d_in[15]; const float* bt = (const float*)d_in[16];
    const float* Wg = (const float*)d_in[17]; const float* bg = (const float*)d_in[18];
    float* out = (float*)d_out;
    char* ws = (char*)d_ws;

    __bf16* Xbf    = (__bf16*)(ws);                    // 20544*512*2  = 21,037,056
    __bf16* Tbf    = (__bf16*)(ws + 21037056);         // 21,037,056
    __bf16* Rbf    = (__bf16*)(ws + 42074112);         // 20544*704*2  = 28,925,952
    __bf16* Bs16   = (__bf16*)(ws + 71000064);         // 96*512*2 = 98,304
    __bf16* Btm16  = (__bf16*)(ws + 71098368);
    __bf16* Bg16   = (__bf16*)(ws + 71196672);
    __bf16* Bst16  = (__bf16*)(ws + 71294976);         // 96*704*2 = 135,168
    __bf16* Bgr16  = (__bf16*)(ws + 71430144);
    __bf16* Bqkv16 = (__bf16*)(ws + 71565312);         // 3*64*64*2 = 24,576; end ~71.6 MB

    k_wprep<<<dim3((PREDn * 704 + 255) / 256), dim3(256), 0, stream>>>(
        Ws, Wt, Wg, Wq, Wk, Wv, Bs16, Btm16, Bg16, Bst16, Bgr16, Bqkv16, out);
    k_decomp<<<dim3(Bn, 3), dim3(128), 0, stream>>>(x_enc, Xbf, Tbf);
    k_attn_tf<<<dim3(Mrows / 2), dim3(32), 0, stream>>>(
        idx, emb, x_dec, te0, te1, te2, Bqkv16, bq, bk, bv, Rbf);
    k_gemm<<<dim3(Mrows / 16, PREDn / 32), dim3(32), 0, stream>>>(
        Xbf, Tbf, Rbf, Bs16, Btm16, Bg16, Bst16, Bgr16, bs, bt, bg, out);
}
